// MambaEncoder_36850819400673
// MI455X (gfx1250) — compile-verified, matrix operand order checked by reading, not run on hardware
//
#include <hip/hip_runtime.h>

typedef __bf16 bf16_t;
typedef __attribute__((ext_vector_type(16))) __bf16 v16bf;
typedef __attribute__((ext_vector_type(8)))  __bf16 v8bf;
typedef __attribute__((ext_vector_type(8)))  float  v8f;

#define D_MODEL  512
#define D_STATE  16
#define N_LAYERS 6
#define D_INNER  1024
#define D_CONV   4
#define DT_RANK  32
#define BB       8
#define LLEN     2048
#define TT       (BB * LLEN)   // 16384 tokens

#if __has_builtin(__builtin_amdgcn_global_load_async_to_lds_b128) && \
    __has_builtin(__builtin_amdgcn_s_wait_asynccnt)
#define USE_ASYNC_LDS 1
typedef int v4i_vs __attribute__((vector_size(16)));
typedef __attribute__((address_space(1))) v4i_vs* g_b128_t;   // global src operand
typedef __attribute__((address_space(3))) v4i_vs* l_b128_t;   // LDS dst operand
#else
#define USE_ASYNC_LDS 0
#endif

// ---------------------------------------------------------------- utilities
__global__ void copy_f32_kernel(const float* __restrict__ in, float* __restrict__ out, int n) {
    int i = blockIdx.x * blockDim.x + threadIdx.x;
    if (i < n) out[i] = in[i];
}

__global__ void cvt_f32_bf16_kernel(const float* __restrict__ in, bf16_t* __restrict__ out, int n) {
    int i = blockIdx.x * blockDim.x + threadIdx.x;
    if (i < n) out[i] = (bf16_t)in[i];
}

// ------------------------------------------------- RMSNorm (+pos) -> bf16
__global__ __launch_bounds__(256) void rmsnorm_pos_kernel(
    const float* __restrict__ x, const float* __restrict__ w,
    const float* __restrict__ pos, bf16_t* __restrict__ u) {
    int t = blockIdx.x;                       // token 0..TT-1
    const float* xr = x   + (size_t)t * D_MODEL;
    const float* pr = pos + (size_t)t * D_MODEL;
    __shared__ float red[256];
    float s = 0.f;
    for (int c = threadIdx.x; c < D_MODEL; c += 256) { float v = xr[c]; s += v * v; }
    red[threadIdx.x] = s; __syncthreads();
    for (int o = 128; o > 0; o >>= 1) {
        if (threadIdx.x < o) red[threadIdx.x] += red[threadIdx.x + o];
        __syncthreads();
    }
    float r = rsqrtf(red[0] * (1.f / D_MODEL) + 1.1920929e-07f);
    for (int c = threadIdx.x; c < D_MODEL; c += 256)
        u[(size_t)t * D_MODEL + c] = (bf16_t)(xr[c] * r * w[c] + pr[c]);
}

__global__ __launch_bounds__(256) void rmsnorm_final_kernel(
    const float* __restrict__ x, const float* __restrict__ w, float* __restrict__ out) {
    int t = blockIdx.x;
    const float* xr = x + (size_t)t * D_MODEL;
    __shared__ float red[256];
    float s = 0.f;
    for (int c = threadIdx.x; c < D_MODEL; c += 256) { float v = xr[c]; s += v * v; }
    red[threadIdx.x] = s; __syncthreads();
    for (int o = 128; o > 0; o >>= 1) {
        if (threadIdx.x < o) red[threadIdx.x] += red[threadIdx.x + o];
        __syncthreads();
    }
    float r = rsqrtf(red[0] * (1.f / D_MODEL) + 1.1920929e-07f);
    for (int c = threadIdx.x; c < D_MODEL; c += 256)
        out[(size_t)t * D_MODEL + c] = xr[c] * r * w[c];
}

// ------------------------------------------------- bf16 WMMA GEMM
// C[M,N] (f32) = A[M,K] (bf16, row-major) * W[N,K]^T (bf16, row-major, K contiguous)
// Block tile 128x64, K-step 32; 8 waves, each wave a 32x32 tile (2x2 WMMA 16x16x32).
// A tile double-buffered in LDS; staged via async global->LDS copies when available.
__global__ __launch_bounds__(256) void gemm_bf16_wmma(
    const bf16_t* __restrict__ A, const bf16_t* __restrict__ W,
    float* __restrict__ C, int M, int N, int K) {
    __shared__ __attribute__((aligned(16))) bf16_t As[2][128][40]; // pad->16B-aligned, conflict-free

    const int tid   = threadIdx.x;
    const int lane  = tid & 31;
    const int wid   = tid >> 5;        // 0..7
    const int wr    = wid >> 1;        // 0..3 (row group of 32)
    const int wc    = wid & 1;         // 0..1 (col group of 32)
    const int m0    = blockIdx.y * 128;
    const int n0    = blockIdx.x * 64;
    const int mhalf = lane >> 4;       // 0/1
    const int l15   = lane & 15;
    const int kb    = mhalf * 8;       // K sub-chunk base per ISA 16-bit A layout

    // each thread owns 2 chunks of 8 bf16 (16B) of the 128x32 tile
    const int r0 = tid >> 2,         c0 = (tid & 3) << 3;
    const int r1 = (tid + 256) >> 2, c1 = (tid & 3) << 3;

    v8f acc[2][2] = {};

    auto stage = [&](int buf, int kk) {
        const bf16_t* g0 = &A[(size_t)(m0 + r0) * K + kk + c0];
        const bf16_t* g1 = &A[(size_t)(m0 + r1) * K + kk + c1];
#if USE_ASYNC_LDS
        __builtin_amdgcn_global_load_async_to_lds_b128(
            (g_b128_t)g0, (l_b128_t)&As[buf][r0][c0], 0, 0);
        __builtin_amdgcn_global_load_async_to_lds_b128(
            (g_b128_t)g1, (l_b128_t)&As[buf][r1][c1], 0, 0);
#else
        *(v8bf*)&As[buf][r0][c0] = *(const v8bf*)g0;
        *(v8bf*)&As[buf][r1][c1] = *(const v8bf*)g1;
#endif
    };

    const int nk = K >> 5;
    stage(0, 0);
#if USE_ASYNC_LDS
    __builtin_amdgcn_s_wait_asynccnt(0);
#endif
    __syncthreads();

    for (int ks = 0; ks < nk; ++ks) {
        const int cur = ks & 1;
        const int kk  = ks << 5;
        if (ks + 1 < nk) {
            stage(cur ^ 1, kk + 32);                   // overlap next A tile with compute
            __builtin_prefetch(&W[(size_t)(n0 + (tid & 63)) * K + kk + 32], 0, 2);
        }

#pragma unroll
        for (int am = 0; am < 2; ++am) {
            v16bf afrag;
            {
                int row = wr * 32 + am * 16 + l15;
                v8bf lo = *(const v8bf*)&As[cur][row][kb];
                v8bf hi = *(const v8bf*)&As[cur][row][16 + kb];
#pragma unroll
                for (int e = 0; e < 8; ++e) { afrag[e] = lo[e]; afrag[e + 8] = hi[e]; }
            }
#pragma unroll
            for (int bn = 0; bn < 2; ++bn) {
                int ncol = n0 + wc * 32 + bn * 16 + l15;
                const bf16_t* wrow = &W[(size_t)ncol * K + kk + kb];
                v8bf blo = *(const v8bf*)wrow;
                v8bf bhi = *(const v8bf*)(wrow + 16);
                v16bf bfrag;
#pragma unroll
                for (int e = 0; e < 8; ++e) { bfrag[e] = blo[e]; bfrag[e + 8] = bhi[e]; }
                acc[am][bn] = __builtin_amdgcn_wmma_f32_16x16x32_bf16(
                    false, afrag, false, bfrag, (short)0, acc[am][bn], false, false);
            }
        }
#if USE_ASYNC_LDS
        __builtin_amdgcn_s_wait_asynccnt(0);           // next tile fully in LDS
#endif
        __syncthreads();
    }

    // Store per ISA 32-bit C/D layout: lane half selects M+0/M+8, VGPR j -> row+j
#pragma unroll
    for (int am = 0; am < 2; ++am) {
        int rbase = m0 + wr * 32 + am * 16 + mhalf * 8;
#pragma unroll
        for (int bn = 0; bn < 2; ++bn) {
            int col = n0 + wc * 32 + bn * 16 + l15;
#pragma unroll
            for (int j = 0; j < 8; ++j)
                C[(size_t)(rbase + j) * N + col] = acc[am][bn][j];
        }
    }
}

// ------------------------------------------------- causal depthwise conv + SiLU
__global__ void conv_silu_kernel(const float* __restrict__ xz, const float* __restrict__ cw,
                                 const float* __restrict__ cb, float* __restrict__ xcs,
                                 bf16_t* __restrict__ xcsb) {
    int id = blockIdx.x * blockDim.x + threadIdx.x;   // over TT*D_INNER
    if (id >= TT * D_INNER) return;
    int d = id & (D_INNER - 1);
    int t = id / D_INNER;
    int l = t & (LLEN - 1);
    float acc = cb[d];
#pragma unroll
    for (int j = 0; j < D_CONV; ++j) {
        int lj = l - (D_CONV - 1) + j;
        if (lj >= 0)
            acc += cw[d * D_CONV + j] * xz[(size_t)(t - (D_CONV - 1) + j) * (2 * D_INNER) + d];
    }
    float s = acc / (1.f + __expf(-acc));
    xcs[id]  = s;
    xcsb[id] = (bf16_t)s;
}

// ------------------------------------------------- dt slice -> bf16
__global__ void slice_dt_kernel(const float* __restrict__ xdbl, bf16_t* __restrict__ dtlow, int n) {
    int id = blockIdx.x * blockDim.x + threadIdx.x;   // over TT*DT_RANK
    if (id >= n) return;
    int t = id / DT_RANK, c = id & (DT_RANK - 1);
    dtlow[id] = (bf16_t)xdbl[(size_t)t * 64 + c];
}

// ------------------------------------------------- softplus(dt + bias)
__global__ void softplus_kernel(float* __restrict__ dt, const float* __restrict__ dtb, int n) {
    int id = blockIdx.x * blockDim.x + threadIdx.x;
    if (id >= n) return;
    int d = id & (D_INNER - 1);
    float v = dt[id] + dtb[d];
    dt[id] = (v > 20.f) ? v : log1pf(__expf(v));
}

// ------------------------------------------------- selective scan
// One thread per (b, channel); h[16], A[16] in registers; B_t/C_t staged in LDS.
__global__ __launch_bounds__(256) void scan_kernel(
    const float* __restrict__ dt, const float* __restrict__ xdbl,
    const float* __restrict__ xcs, const float* __restrict__ a_log,
    float* __restrict__ y) {
    int b = blockIdx.x >> 2;
    int d = ((blockIdx.x & 3) << 8) + threadIdx.x;
    float A[D_STATE], h[D_STATE];
#pragma unroll
    for (int n = 0; n < D_STATE; ++n) { A[n] = -__expf(a_log[d * D_STATE + n]); h[n] = 0.f; }
    __shared__ float sBC[32];
    for (int l = 0; l < LLEN; ++l) {
        size_t t = (size_t)b * LLEN + l;
        if (threadIdx.x < 32) sBC[threadIdx.x] = xdbl[t * 64 + 32 + threadIdx.x];
        __syncthreads();
        float dtv = dt[t * D_INNER + d];
        float u   = xcs[t * D_INNER + d];
        float du  = dtv * u;
        float yv  = 0.f;
#pragma unroll
        for (int n = 0; n < D_STATE; ++n) {
            float dA = __expf(dtv * A[n]);
            h[n] = h[n] * dA + du * sBC[n];
            yv += h[n] * sBC[16 + n];
        }
        y[t * D_INNER + d] = yv;
        __syncthreads();
    }
}

// ------------------------------------------------- (y + D*x) * SiLU(z) -> bf16
__global__ void gate_kernel(const float* __restrict__ y, const float* __restrict__ xcs,
                            const float* __restrict__ xz, const float* __restrict__ Dp,
                            bf16_t* __restrict__ ygb, int n) {
    int id = blockIdx.x * blockDim.x + threadIdx.x;
    if (id >= n) return;
    int d = id & (D_INNER - 1);
    int t = id / D_INNER;
    float z  = xz[(size_t)t * (2 * D_INNER) + D_INNER + d];
    float yv = y[id] + Dp[d] * xcs[id];
    float g  = z / (1.f + __expf(-z));
    ygb[id] = (bf16_t)(yv * g);
}

// ------------------------------------------------- masked residual add
__global__ void residual_kernel(float* __restrict__ xres, const float* __restrict__ out,
                                const int* __restrict__ mask, int n) {
    int id = blockIdx.x * blockDim.x + threadIdx.x;
    if (id >= n) return;
    int t = id / D_MODEL;
    xres[id] += out[id] * (float)mask[t];
}

// ================================================================= launcher
extern "C" void kernel_launch(void* const* d_in, const int* in_sizes, int n_in,
                              void* d_out, int out_size, void* d_ws, size_t ws_size,
                              hipStream_t stream) {
    const float* x        = (const float*)d_in[0];
    const float* pos      = (const float*)d_in[1];
    const int*   mask     = (const int*)d_in[2];
    const float* norm_w   = (const float*)d_in[3];
    const float* fnorm_w  = (const float*)d_in[4];
    const float* in_w     = (const float*)d_in[5];
    const float* conv_w   = (const float*)d_in[6];
    const float* conv_b   = (const float*)d_in[7];
    const float* xp_w     = (const float*)d_in[8];
    const float* dt_w     = (const float*)d_in[9];
    const float* dt_b     = (const float*)d_in[10];
    const float* A_log    = (const float*)d_in[11];
    const float* D_par    = (const float*)d_in[12];
    const float* out_w    = (const float*)d_in[13];
    float* out = (float*)d_out;

    char* ws = (char*)d_ws;
    size_t off = 0;
    auto alloc = [&](size_t bytes) -> char* {
        char* p = ws + off; off += (bytes + 255) & ~(size_t)255; return p;
    };
    float*  xres    = (float*) alloc((size_t)TT * D_MODEL * 4);
    bf16_t* u_b     = (bf16_t*)alloc((size_t)TT * D_MODEL * 2);
    float*  xz      = (float*) alloc((size_t)TT * 2 * D_INNER * 4);
    float*  xcs     = (float*) alloc((size_t)TT * D_INNER * 4);
    bf16_t* xcs_b   = (bf16_t*)alloc((size_t)TT * D_INNER * 2);
    float*  xdbl    = (float*) alloc((size_t)TT * 64 * 4);
    bf16_t* dtlow_b = (bf16_t*)alloc((size_t)TT * DT_RANK * 2);
    float*  dtbuf   = (float*) alloc((size_t)TT * D_INNER * 4);
    float*  ybuf    = (float*) alloc((size_t)TT * D_INNER * 4);
    bf16_t* yg_b    = (bf16_t*)alloc((size_t)TT * D_INNER * 2);
    float*  outbuf  = (float*) alloc((size_t)TT * D_MODEL * 4);
    bf16_t* inW_b   = (bf16_t*)alloc((size_t)N_LAYERS * 2 * D_INNER * D_MODEL * 2);
    bf16_t* xpW_b   = (bf16_t*)alloc((size_t)N_LAYERS * 64 * D_INNER * 2);
    bf16_t* dtW_b   = (bf16_t*)alloc((size_t)N_LAYERS * D_INNER * DT_RANK * 2);
    bf16_t* outW_b  = (bf16_t*)alloc((size_t)N_LAYERS * D_MODEL * D_INNER * 2);

    int n;
    n = TT * D_MODEL;
    copy_f32_kernel<<<(n + 255) / 256, 256, 0, stream>>>(x, xres, n);
    n = N_LAYERS * 2 * D_INNER * D_MODEL;
    cvt_f32_bf16_kernel<<<(n + 255) / 256, 256, 0, stream>>>(in_w, inW_b, n);
    n = N_LAYERS * 64 * D_INNER;
    cvt_f32_bf16_kernel<<<(n + 255) / 256, 256, 0, stream>>>(xp_w, xpW_b, n);
    n = N_LAYERS * D_INNER * DT_RANK;
    cvt_f32_bf16_kernel<<<(n + 255) / 256, 256, 0, stream>>>(dt_w, dtW_b, n);
    n = N_LAYERS * D_MODEL * D_INNER;
    cvt_f32_bf16_kernel<<<(n + 255) / 256, 256, 0, stream>>>(out_w, outW_b, n);

    for (int i = 0; i < N_LAYERS; ++i) {
        rmsnorm_pos_kernel<<<TT, 256, 0, stream>>>(xres, norm_w + i * D_MODEL, pos, u_b);

        gemm_bf16_wmma<<<dim3((2 * D_INNER) / 64, TT / 128), 256, 0, stream>>>(
            u_b, inW_b + (size_t)i * 2 * D_INNER * D_MODEL, xz, TT, 2 * D_INNER, D_MODEL);

        n = TT * D_INNER;
        conv_silu_kernel<<<(n + 255) / 256, 256, 0, stream>>>(
            xz, conv_w + i * D_INNER * D_CONV, conv_b + i * D_INNER, xcs, xcs_b);

        gemm_bf16_wmma<<<dim3(64 / 64, TT / 128), 256, 0, stream>>>(
            xcs_b, xpW_b + (size_t)i * 64 * D_INNER, xdbl, TT, 64, D_INNER);

        n = TT * DT_RANK;
        slice_dt_kernel<<<(n + 255) / 256, 256, 0, stream>>>(xdbl, dtlow_b, n);

        gemm_bf16_wmma<<<dim3(D_INNER / 64, TT / 128), 256, 0, stream>>>(
            dtlow_b, dtW_b + (size_t)i * D_INNER * DT_RANK, dtbuf, TT, D_INNER, DT_RANK);

        n = TT * D_INNER;
        softplus_kernel<<<(n + 255) / 256, 256, 0, stream>>>(dtbuf, dt_b + i * D_INNER, n);

        scan_kernel<<<BB * (D_INNER / 256), 256, 0, stream>>>(
            dtbuf, xdbl, xcs, A_log + (size_t)i * D_INNER * D_STATE, ybuf);

        gate_kernel<<<(n + 255) / 256, 256, 0, stream>>>(
            ybuf, xcs, xz, D_par + i * D_INNER, yg_b, n);

        gemm_bf16_wmma<<<dim3(D_MODEL / 64, TT / 128), 256, 0, stream>>>(
            yg_b, outW_b + (size_t)i * D_MODEL * D_INNER, outbuf, TT, D_MODEL, D_INNER);

        n = TT * D_MODEL;
        residual_kernel<<<(n + 255) / 256, 256, 0, stream>>>(xres, outbuf, mask, n);
    }

    rmsnorm_final_kernel<<<TT, 256, 0, stream>>>(xres, fnorm_w, out);
}
